// KNNTransformer_46875273068856
// MI455X (gfx1250) — compile-verified, matrix operand order checked by reading, not run on hardware
//
#include <hip/hip_runtime.h>
#include <hip/hip_bf16.h>
#include <math.h>

#define NPOINTS 50000
#define NGROUPS 2048
#define GSIZE   32
#define FEAT    6
#define HID     128
#define NDEPTH  16
#define NHEADS  8
#define HDIM    16
#define EPS_F   1e-5f
#define BIGF    1e10f

typedef __attribute__((ext_vector_type(16))) __bf16 v16bf;
typedef __attribute__((ext_vector_type(8)))  float  v8f;

// ---------------------------------------------------------------------------
// WMMA fragment helpers (bf16 16x16x32, wave32 layouts per CDNA5 ISA 7.12.2)
// A-frag / B^T-frag: lane L<16 holds row L; elem e -> k = k0 + 8g + e (+8 if e>=8)
// ---------------------------------------------------------------------------

// Generic masked scalar loader (edge cases: K=6, N=8)
__device__ __forceinline__ v16bf load_frag(const float* base, int sR, int sK,
                                           int r0, int k0, int rMax, int kMax,
                                           const float* ascale, const float* ashift) {
  int lane = threadIdx.x & 31;
  int r = r0 + (lane & 15);
  int g = (lane >> 4) & 1;
  bool rOK = (r < rMax);
  v16bf out;
#pragma unroll
  for (int e = 0; e < 16; ++e) {
    int k = k0 + e + 8 * g + ((e >= 8) ? 8 : 0);
    float v = 0.0f;
    if (rOK && k < kMax) {
      v = base[(size_t)r * sR + (size_t)k * sK];
      if (ascale) v = fmaxf(v * ascale[k] + ashift[k], 0.0f);   // fused BN+ReLU
    }
    out[e] = (__bf16)v;
  }
  return out;
}

// Fast path: row-major, contiguous k, 32 valid k's, row in range. 4x float4.
__device__ __forceinline__ v16bf load_frag_vec(const float* base, int sR,
                                               int r0, int k0,
                                               const float* ascale, const float* ashift) {
  int lane = threadIdx.x & 31;
  int r = r0 + (lane & 15);
  int g = (lane >> 4) & 1;
  int kb = k0 + 8 * g;
  const float* p = base + (size_t)r * sR + kb;
  float4 f0 = *(const float4*)(p);
  float4 f1 = *(const float4*)(p + 4);
  float4 f2 = *(const float4*)(p + 16);
  float4 f3 = *(const float4*)(p + 20);
  float vv[16] = {f0.x, f0.y, f0.z, f0.w, f1.x, f1.y, f1.z, f1.w,
                  f2.x, f2.y, f2.z, f2.w, f3.x, f3.y, f3.z, f3.w};
  v16bf out;
  if (ascale) {
#pragma unroll
    for (int e = 0; e < 16; ++e) {
      int k = kb + e + ((e >= 8) ? 8 : 0);
      out[e] = (__bf16)fmaxf(vv[e] * ascale[k] + ashift[k], 0.0f);
    }
  } else {
#pragma unroll
    for (int e = 0; e < 16; ++e) out[e] = (__bf16)vv[e];
  }
  return out;
}

// 16 valid k's (k 0..15), upper half of contraction zero-padded. 2x float4.
__device__ __forceinline__ v16bf load_frag_k16(const float* base, int sR, int r0) {
  int lane = threadIdx.x & 31;
  int r = r0 + (lane & 15);
  int g = (lane >> 4) & 1;
  const float* p = base + (size_t)r * sR + 8 * g;
  float4 f0 = *(const float4*)(p);
  float4 f1 = *(const float4*)(p + 4);
  float vv[8] = {f0.x, f0.y, f0.z, f0.w, f1.x, f1.y, f1.z, f1.w};
  v16bf out;
#pragma unroll
  for (int e = 0; e < 8; ++e) out[e] = (__bf16)vv[e];
#pragma unroll
  for (int e = 8; e < 16; ++e) out[e] = (__bf16)0.0f;
  return out;
}

// Build padded fragment from 8 consecutive floats (LDS)
__device__ __forceinline__ v16bf frag8(const float* p) {
  float4 f0 = *(const float4*)(p);
  float4 f1 = *(const float4*)(p + 4);
  float vv[8] = {f0.x, f0.y, f0.z, f0.w, f1.x, f1.y, f1.z, f1.w};
  v16bf out;
#pragma unroll
  for (int e = 0; e < 8; ++e) out[e] = (__bf16)vv[e];
#pragma unroll
  for (int e = 8; e < 16; ++e) out[e] = (__bf16)0.0f;
  return out;
}

// V^T fragment: row n = head-dim d, k = key (strided), padded to 32
__device__ __forceinline__ v16bf load_vfrag(const float* base, int sK) {
  int lane = threadIdx.x & 31;
  int d = lane & 15;
  int g = (lane >> 4) & 1;
  v16bf out;
#pragma unroll
  for (int e = 0; e < 8; ++e) out[e] = (__bf16)base[(size_t)(e + 8 * g) * sK + d];
#pragma unroll
  for (int e = 8; e < 16; ++e) out[e] = (__bf16)0.0f;
  return out;
}

// ---------------------------------------------------------------------------
// GEMM: C = act(A @ W^T + bias) (+resid). A:MxK, W:NxK row-major.
// 128 threads = 4 waves; each wave owns a 64x16 strip (4 acc share one B-frag).
// ---------------------------------------------------------------------------
template <int ACT>
__global__ void gemm_kernel(const float* __restrict__ A, const float* __restrict__ W,
                            const float* __restrict__ bias, const float* __restrict__ resid,
                            float* __restrict__ C, int M, int N, int K,
                            const float* __restrict__ ascale, const float* __restrict__ ashift) {
  int wave = threadIdx.x >> 5;
  int tn = blockIdx.x * 4 + wave;
  int c0 = tn * 16;
  int r0 = blockIdx.y * 64;
  if (c0 >= N || r0 >= M) return;

  bool vec_ok = ((K & 31) == 0);
  bool vecB = vec_ok && (c0 + 15 < N);
  bool vecA = vec_ok && (r0 + 63 < M);

  v8f acc0 = {}, acc1 = {}, acc2 = {}, acc3 = {};
  for (int k = 0; k < K; k += 32) {
    __builtin_prefetch(A + (size_t)r0 * K + k + 64, 0, 1);
    v16bf b = vecB ? load_frag_vec(W, K, c0, k, nullptr, nullptr)
                   : load_frag(W, K, 1, c0, k, N, K, nullptr, nullptr);
    v16bf a0 = vecA ? load_frag_vec(A, K, r0, k, ascale, ashift)
                    : load_frag(A, K, 1, r0, k, M, K, ascale, ashift);
    acc0 = __builtin_amdgcn_wmma_f32_16x16x32_bf16(false, a0, false, b, (short)0, acc0, false, false);
    v16bf a1 = vecA ? load_frag_vec(A, K, r0 + 16, k, ascale, ashift)
                    : load_frag(A, K, 1, r0 + 16, k, M, K, ascale, ashift);
    acc1 = __builtin_amdgcn_wmma_f32_16x16x32_bf16(false, a1, false, b, (short)0, acc1, false, false);
    v16bf a2 = vecA ? load_frag_vec(A, K, r0 + 32, k, ascale, ashift)
                    : load_frag(A, K, 1, r0 + 32, k, M, K, ascale, ashift);
    acc2 = __builtin_amdgcn_wmma_f32_16x16x32_bf16(false, a2, false, b, (short)0, acc2, false, false);
    v16bf a3 = vecA ? load_frag_vec(A, K, r0 + 48, k, ascale, ashift)
                    : load_frag(A, K, 1, r0 + 48, k, M, K, ascale, ashift);
    acc3 = __builtin_amdgcn_wmma_f32_16x16x32_bf16(false, a3, false, b, (short)0, acc3, false, false);
  }

  int lane = threadIdx.x & 31;
  int c = c0 + (lane & 15);
  int hi = (lane >> 4) & 1;
  float bv = (c < N) ? bias[c] : 0.0f;
  v8f accs[4] = {acc0, acc1, acc2, acc3};
#pragma unroll
  for (int t = 0; t < 4; ++t) {
#pragma unroll
    for (int e = 0; e < 8; ++e) {
      int r = r0 + 16 * t + e + 8 * hi;
      if (r < M && c < N) {
        float v = accs[t][e] + bv;
        if (ACT == 1) v = 0.5f * v * (1.0f + erff(v * 0.70710678118654752f));
        if (resid) v += resid[(size_t)r * N + c];
        C[(size_t)r * N + c] = v;
      }
    }
  }
}

// ---------------------------------------------------------------------------
// Flash attention: one wave per (head, 16-query tile). Row-owner softmax:
// lanes 0..15 own one row each; no cross-lane permutes, LDS float4 traffic only.
// ---------------------------------------------------------------------------
__global__ void attn_kernel(const float* __restrict__ qkv, float* __restrict__ out, int G) {
  const int h  = blockIdx.x;
  const int q0 = blockIdx.y * 16;
  const int lane = threadIdx.x & 31;
  const int col = lane & 15;
  const int hi  = (lane >> 4) & 1;
  const int LD  = 3 * HID;           // 384
  const float scale = 0.25f;         // HDIM^-0.5

  __shared__ __align__(16) float sS[16 * 16];   // S tile, then P tile
  __shared__ __align__(16) float sAux[16];      // alpha, then l

  v16bf qa = load_frag_k16(qkv + h * HDIM, LD, q0);

  float mrow = -1.0e30f, lrow = 0.0f;  // valid in lanes 0..15 only
  v8f o = {};

  for (int kt = 0; kt < G; kt += 16) {
    v16bf kf = load_frag_k16(qkv + HID + h * HDIM, LD, kt);
    v8f s = {};
    s = __builtin_amdgcn_wmma_f32_16x16x32_bf16(false, qa, false, kf, (short)0, s, false, false);
#pragma unroll
    for (int e = 0; e < 8; ++e) sS[(e + 8 * hi) * 16 + col] = s[e] * scale;
    __syncthreads();

    if (lane < 16) {
      const float4* rp = (const float4*)&sS[lane * 16];
      float4 a0 = rp[0], a1 = rp[1], a2 = rp[2], a3 = rp[3];
      float sv[16] = {a0.x, a0.y, a0.z, a0.w, a1.x, a1.y, a1.z, a1.w,
                      a2.x, a2.y, a2.z, a2.w, a3.x, a3.y, a3.z, a3.w};
      float mx = sv[0];
#pragma unroll
      for (int j = 1; j < 16; ++j) mx = fmaxf(mx, sv[j]);
      float mn = fmaxf(mrow, mx);
      float p[16]; float sum = 0.0f;
#pragma unroll
      for (int j = 0; j < 16; ++j) { p[j] = __expf(sv[j] - mn); sum += p[j]; }
      float alpha = __expf(mrow - mn);
      lrow = lrow * alpha + sum;
      mrow = mn;
      float4* wp = (float4*)&sS[lane * 16];
      wp[0] = make_float4(p[0], p[1], p[2], p[3]);
      wp[1] = make_float4(p[4], p[5], p[6], p[7]);
      wp[2] = make_float4(p[8], p[9], p[10], p[11]);
      wp[3] = make_float4(p[12], p[13], p[14], p[15]);
      sAux[lane] = alpha;
    }
    __syncthreads();

    const float4* ap = (const float4*)&sAux[8 * hi];
    float4 A0 = ap[0], A1 = ap[1];
    o[0] *= A0.x; o[1] *= A0.y; o[2] *= A0.z; o[3] *= A0.w;
    o[4] *= A1.x; o[5] *= A1.y; o[6] *= A1.z; o[7] *= A1.w;

    v16bf pf = frag8(&sS[col * 16 + 8 * hi]);
    v16bf vf = load_vfrag(qkv + (size_t)kt * LD + 2 * HID + h * HDIM, LD);
    o = __builtin_amdgcn_wmma_f32_16x16x32_bf16(false, pf, false, vf, (short)0, o, false, false);
    __syncthreads();
  }

  if (lane < 16) sAux[lane] = lrow;
  __syncthreads();
  const float4* lp = (const float4*)&sAux[8 * hi];
  float4 L0 = lp[0], L1 = lp[1];
  float li[8] = {L0.x, L0.y, L0.z, L0.w, L1.x, L1.y, L1.z, L1.w};
#pragma unroll
  for (int e = 0; e < 8; ++e) {
    int r = q0 + e + 8 * hi;
    out[(size_t)r * HID + h * HDIM + col] = o[e] / li[e];
  }
}

// ---------------------------------------------------------------------------
// LayerNorm over rows of width 128 (wave per row, 8 rows / block)
// ---------------------------------------------------------------------------
__global__ void ln_kernel(const float* __restrict__ x, const float* __restrict__ g,
                          const float* __restrict__ b, float* __restrict__ y, int M) {
  int row = blockIdx.x * 8 + (threadIdx.x >> 5);
  int lane = threadIdx.x & 31;
  if (row >= M) return;
  float v[4]; float s = 0.0f;
#pragma unroll
  for (int i = 0; i < 4; ++i) { v[i] = x[(size_t)row * HID + lane + 32 * i]; s += v[i]; }
#pragma unroll
  for (int off = 16; off > 0; off >>= 1) s += __shfl_xor(s, off, 32);
  float mean = s * (1.0f / HID);
  float var = 0.0f;
#pragma unroll
  for (int i = 0; i < 4; ++i) { float d = v[i] - mean; var += d * d; }
#pragma unroll
  for (int off = 16; off > 0; off >>= 1) var += __shfl_xor(var, off, 32);
  var *= (1.0f / HID);
  float rs = rsqrtf(var + EPS_F);
#pragma unroll
  for (int i = 0; i < 4; ++i) {
    int c = lane + 32 * i;
    y[(size_t)row * HID + c] = (v[i] - mean) * rs * g[c] + b[c];
  }
}

// ---------------------------------------------------------------------------
// BatchNorm helpers
// ---------------------------------------------------------------------------
__global__ void bn_stats_kernel(const float* __restrict__ x, float* mean, float* var_,
                                int M, int N) {
  int c = blockIdx.x, tid = threadIdx.x;
  float s = 0.0f, s2 = 0.0f;
  for (int r = tid; r < M; r += blockDim.x) {
    float v = x[(size_t)r * N + c];
    s += v; s2 += v * v;
  }
  __shared__ float sh1[256], sh2[256];
  sh1[tid] = s; sh2[tid] = s2; __syncthreads();
  for (int st = 128; st > 0; st >>= 1) {
    if (tid < st) { sh1[tid] += sh1[tid + st]; sh2[tid] += sh2[tid + st]; }
    __syncthreads();
  }
  if (tid == 0) {
    float m = sh1[0] / M;
    mean[c] = m;
    var_[c] = sh2[0] / M - m * m;
  }
}

__global__ void bn_coef_kernel(const float* mean, const float* var_, const float* g,
                               const float* b, float* scale, float* shift, int N) {
  int c = blockIdx.x * blockDim.x + threadIdx.x;
  if (c >= N) return;
  float sc = g[c] * rsqrtf(var_[c] + EPS_F);
  scale[c] = sc;
  shift[c] = b[c] - mean[c] * sc;
}

// ---------------------------------------------------------------------------
// FPS: one persistent block, 2048 sequential steps; wave-level reduce then
// one cross-wave reduce -> only 2 barriers per step.
// ---------------------------------------------------------------------------
__global__ void fps_kernel(const float* __restrict__ x, float* __restrict__ dist,
                           float* __restrict__ cx, int* __restrict__ cidx) {
  __shared__ float swm[32];
  __shared__ int   swi[32];
  __shared__ int   sFar;
  int tid = threadIdx.x;
  int lane = tid & 31, wv = tid >> 5, nw = blockDim.x >> 5;
  for (int i = tid; i < NPOINTS; i += blockDim.x) dist[i] = BIGF;
  __syncthreads();
  int far = 0;
  for (int it = 0; it < NGROUPS; ++it) {
    float fx = x[far * 3], fy = x[far * 3 + 1], fz = x[far * 3 + 2];
    if (tid == 0) {
      cidx[it] = far;
      cx[it * 3] = fx; cx[it * 3 + 1] = fy; cx[it * 3 + 2] = fz;
    }
    float lm = -1.0f; int li = 0;
    for (int i = tid; i < NPOINTS; i += blockDim.x) {
      float dx = x[i * 3] - fx, dy = x[i * 3 + 1] - fy, dz = x[i * 3 + 2] - fz;
      float nd = fminf(dist[i], dx * dx + dy * dy + dz * dz);
      dist[i] = nd;
      if (nd > lm) { lm = nd; li = i; }
    }
#pragma unroll
    for (int off = 16; off > 0; off >>= 1) {
      float om = __shfl_xor(lm, off, 32);
      int   oi = __shfl_xor(li, off, 32);
      if (om > lm || (om == lm && oi < li)) { lm = om; li = oi; }
    }
    if (lane == 0) { swm[wv] = lm; swi[wv] = li; }
    __syncthreads();
    if (wv == 0) {
      float am = (lane < nw) ? swm[lane] : -2.0f;
      int   ai = (lane < nw) ? swi[lane] : 0;
#pragma unroll
      for (int off = 16; off > 0; off >>= 1) {
        float om = __shfl_xor(am, off, 32);
        int   oi = __shfl_xor(ai, off, 32);
        if (om > am || (om == am && oi < ai)) { am = om; ai = oi; }
      }
      if (lane == 0) sFar = ai;
    }
    __syncthreads();
    far = sFar;
  }
}

// ---------------------------------------------------------------------------
// Per-point nearest center (argmin over 2048 centers staged in LDS)
// ---------------------------------------------------------------------------
__global__ void nearest_kernel(const float* __restrict__ x, const float* __restrict__ cx,
                               int* __restrict__ nearest) {
  __shared__ float sc[NGROUPS * 3];
  for (int i = threadIdx.x; i < NGROUPS * 3; i += blockDim.x) sc[i] = cx[i];
  __syncthreads();
  int i = blockIdx.x * blockDim.x + threadIdx.x;
  if (i >= NPOINTS) return;
  float px = x[i * 3], py = x[i * 3 + 1], pz = x[i * 3 + 2];
  float best = 3.0e38f; int bi = 0;
  for (int c = 0; c < NGROUPS; ++c) {
    float dx = sc[c * 3] - px, dy = sc[c * 3 + 1] - py, dz = sc[c * 3 + 2] - pz;
    float d = dx * dx + dy * dy + dz * dz;
    if (d < best) { best = d; bi = c; }
  }
  nearest[i] = bi;
}

// ---------------------------------------------------------------------------
// KNN top-32 per center (wave per center; per-lane top-32 + wave-min merge)
// with fused feature gather into nb[NGROUPS][32][FEAT]
// ---------------------------------------------------------------------------
__global__ void knn_gather_kernel(const float* __restrict__ x, const float* __restrict__ cx,
                                  const float* __restrict__ feats, float* __restrict__ nb) {
  int g = blockIdx.x;
  int lane = threadIdx.x & 31;
  float gx = cx[g * 3], gy = cx[g * 3 + 1], gz = cx[g * 3 + 2];

  float bd[32]; int bi_[32];
#pragma unroll
  for (int j = 0; j < 32; ++j) { bd[j] = 3.0e38f; bi_[j] = 0; }
  float cmax = 3.0e38f; int cpos = 0;

  for (int i = lane; i < NPOINTS; i += 32) {
    float dx = x[i * 3] - gx, dy = x[i * 3 + 1] - gy, dz = x[i * 3 + 2] - gz;
    float d = dx * dx + dy * dy + dz * dz;
    if (d < cmax) {
      bd[cpos] = d; bi_[cpos] = i;
      cmax = bd[0]; cpos = 0;
#pragma unroll
      for (int j = 1; j < 32; ++j)
        if (bd[j] > cmax) { cmax = bd[j]; cpos = j; }
    }
  }

  for (int sel = 0; sel < 32; ++sel) {
    float lmin = bd[0]; int lpos = 0;
#pragma unroll
    for (int j = 1; j < 32; ++j)
      if (bd[j] < lmin) { lmin = bd[j]; lpos = j; }
    float v = lmin; int src = lane, pos = lpos, idx = bi_[lpos];
#pragma unroll
    for (int off = 16; off > 0; off >>= 1) {
      float ov = __shfl_xor(v, off, 32);
      int osrc = __shfl_xor(src, off, 32);
      int opos = __shfl_xor(pos, off, 32);
      int oidx = __shfl_xor(idx, off, 32);
      if (ov < v || (ov == v && oidx < idx)) { v = ov; src = osrc; pos = opos; idx = oidx; }
    }
    if (lane == src) bd[pos] = 3.5e38f;   // consume
    if (lane == sel) {
#pragma unroll
      for (int f = 0; f < FEAT; ++f)
        nb[((size_t)g * GSIZE + sel) * FEAT + f] = feats[(size_t)idx * FEAT + f];
    }
  }
}

// ---------------------------------------------------------------------------
// Elementwise helpers
// ---------------------------------------------------------------------------
__global__ void maxpool_kernel(const float* __restrict__ x, float* __restrict__ y, int C) {
  int g = blockIdx.x, c = threadIdx.x;
  if (c >= C) return;
  float m = -3.0e38f;
  for (int s = 0; s < GSIZE; ++s) m = fmaxf(m, x[((size_t)g * GSIZE + s) * C + c]);
  y[(size_t)g * C + c] = m;
}

__global__ void maxpool_pos_kernel(const float* __restrict__ x, const float* __restrict__ pos,
                                   float* __restrict__ y, int C) {
  int g = blockIdx.x, c = threadIdx.x;
  if (c >= C) return;
  float m = -3.0e38f;
  for (int s = 0; s < GSIZE; ++s) m = fmaxf(m, x[((size_t)g * GSIZE + s) * C + c]);
  y[(size_t)g * C + c] = m + pos[(size_t)g * C + c];
}

__global__ void concat_kernel(const float* __restrict__ h2, const float* __restrict__ gmax,
                              float* __restrict__ hcat, int total, int C) {
  int i = blockIdx.x * blockDim.x + threadIdx.x;
  if (i >= total) return;
  int c = i % C, row = i / C, g = row / GSIZE;
  hcat[(size_t)row * (2 * C) + c] = h2[i];
  hcat[(size_t)row * (2 * C) + C + c] = gmax[(size_t)g * C + c];
}

__global__ void gather_out_kernel(const int* __restrict__ nearest, const float* __restrict__ e,
                                  const float* __restrict__ pc, float* __restrict__ out) {
  int i = blockIdx.x * blockDim.x + threadIdx.x;
  if (i >= NPOINTS) return;
  int c = nearest[i];
#pragma unroll
  for (int j = 0; j < 8; ++j) {
    out[(size_t)i * 8 + j] = e[(size_t)c * 8 + j];
    out[(size_t)NPOINTS * 8 + (size_t)i * 8 + j] = pc[(size_t)c * 8 + j];
  }
}

// ---------------------------------------------------------------------------
// Host orchestration
// ---------------------------------------------------------------------------
static void launch_gemm(const float* A, const float* W, const float* b, const float* resid,
                        float* C, int M, int N, int K, const float* sc, const float* sh,
                        int act, hipStream_t stream) {
  dim3 grid((((N + 15) / 16) + 3) / 4, (M + 63) / 64);
  if (act == 1)
    gemm_kernel<1><<<grid, 128, 0, stream>>>(A, W, b, resid, C, M, N, K, sc, sh);
  else
    gemm_kernel<0><<<grid, 128, 0, stream>>>(A, W, b, resid, C, M, N, K, sc, sh);
}

extern "C" void kernel_launch(void* const* d_in, const int* in_sizes, int n_in,
                              void* d_out, int out_size, void* d_ws, size_t ws_size,
                              hipStream_t stream) {
  const float* x        = (const float*)d_in[0];
  const float* features = (const float*)d_in[1];
  const float* c1_w1 = (const float*)d_in[2];
  const float* c1_b1 = (const float*)d_in[3];
  const float* bn1_g = (const float*)d_in[4];
  const float* bn1_b = (const float*)d_in[5];
  const float* c1_w2 = (const float*)d_in[6];
  const float* c1_b2 = (const float*)d_in[7];
  const float* c2_w1 = (const float*)d_in[8];
  const float* c2_b1 = (const float*)d_in[9];
  const float* bn2_g = (const float*)d_in[10];
  const float* bn2_b = (const float*)d_in[11];
  const float* c2_w2 = (const float*)d_in[12];
  const float* c2_b2 = (const float*)d_in[13];
  const float* pos_emb = (const float*)d_in[14];
  const float* ln1_g = (const float*)d_in[15];
  const float* ln1_b = (const float*)d_in[16];
  const float* qkv_w = (const float*)d_in[17];
  const float* qkv_b = (const float*)d_in[18];
  const float* fc_w  = (const float*)d_in[19];
  const float* fc_b  = (const float*)d_in[20];
  const float* ln2_g = (const float*)d_in[21];
  const float* ln2_b = (const float*)d_in[22];
  const float* mlp_w1 = (const float*)d_in[23];
  const float* mlp_b1 = (const float*)d_in[24];
  const float* mlp_w2 = (const float*)d_in[25];
  const float* mlp_b2 = (const float*)d_in[26];
  const float* he_ln_g = (const float*)d_in[27];
  const float* he_ln_b = (const float*)d_in[28];
  const float* he_w1 = (const float*)d_in[29];
  const float* he_b1 = (const float*)d_in[30];
  const float* he_w2 = (const float*)d_in[31];
  const float* he_b2 = (const float*)d_in[32];
  const float* hp_ln_g = (const float*)d_in[33];
  const float* hp_ln_b = (const float*)d_in[34];
  const float* hp_w1 = (const float*)d_in[35];
  const float* hp_b1 = (const float*)d_in[36];
  const float* hp_w2 = (const float*)d_in[37];
  const float* hp_b2 = (const float*)d_in[38];

  const int ROWS = NGROUPS * GSIZE;  // 65536
  float* wsf = (float*)d_ws;
  size_t off = 0;
  float* dist = wsf + off; off += NPOINTS;
  float* cx   = wsf + off; off += (size_t)NGROUPS * 3;
  int*   cidx = (int*)(wsf + off); off += NGROUPS;
  int*   nearest = (int*)(wsf + off); off += NPOINTS;
  float* nb = wsf + off; off += (size_t)ROWS * FEAT;
  float* B0 = wsf + off; off += (size_t)ROWS * 128;   // h1 -> h3
  float* B1 = wsf + off; off += (size_t)ROWS * 128;   // h2 -> h4
  float* B2 = wsf + off; off += (size_t)ROWS * 128;   // hcat
  float* meanb  = wsf + off; off += 128;
  float* varb   = wsf + off; off += 128;
  float* scaleb = wsf + off; off += 128;
  float* shiftb = wsf + off; off += 128;
  float* gmax = wsf + off; off += (size_t)NGROUPS * 64;
  float* h    = wsf + off; off += (size_t)NGROUPS * HID;
  float* ybuf = wsf + off; off += (size_t)NGROUPS * HID;
  float* qkvb = wsf + off; off += (size_t)NGROUPS * 3 * HID;
  float* attno = wsf + off; off += (size_t)NGROUPS * HID;
  float* gbuf = wsf + off; off += (size_t)NGROUPS * 4 * HID;
  float* t1   = wsf + off; off += (size_t)NGROUPS * 2 * HID;
  float* ebuf = wsf + off; off += (size_t)NGROUPS * 8;
  float* pbuf = wsf + off; off += (size_t)NGROUPS * 8;

  // ---- grouping ----
  fps_kernel<<<1, 1024, 0, stream>>>(x, dist, cx, cidx);
  nearest_kernel<<<(NPOINTS + 255) / 256, 256, 0, stream>>>(x, cx, nearest);
  knn_gather_kernel<<<NGROUPS, 32, 0, stream>>>(x, cx, features, nb);

  // ---- group encoder ----
  launch_gemm(nb, c1_w1, c1_b1, nullptr, B0, ROWS, 32, FEAT, nullptr, nullptr, 0, stream);
  bn_stats_kernel<<<32, 256, 0, stream>>>(B0, meanb, varb, ROWS, 32);
  bn_coef_kernel<<<1, 32, 0, stream>>>(meanb, varb, bn1_g, bn1_b, scaleb, shiftb, 32);
  launch_gemm(B0, c1_w2, c1_b2, nullptr, B1, ROWS, 64, 32, scaleb, shiftb, 0, stream);
  maxpool_kernel<<<NGROUPS, 64, 0, stream>>>(B1, gmax, 64);
  concat_kernel<<<(ROWS * 64 + 255) / 256, 256, 0, stream>>>(B1, gmax, B2, ROWS * 64, 64);
  launch_gemm(B2, c2_w1, c2_b1, nullptr, B0, ROWS, 128, 128, nullptr, nullptr, 0, stream);
  bn_stats_kernel<<<128, 256, 0, stream>>>(B0, meanb, varb, ROWS, 128);
  bn_coef_kernel<<<1, 128, 0, stream>>>(meanb, varb, bn2_g, bn2_b, scaleb, shiftb, 128);
  launch_gemm(B0, c2_w2, c2_b2, nullptr, B1, ROWS, 128, 128, scaleb, shiftb, 0, stream);
  maxpool_pos_kernel<<<NGROUPS, 128, 0, stream>>>(B1, pos_emb, h, 128);

  // ---- transformer ----
  for (int l = 0; l < NDEPTH; ++l) {
    ln_kernel<<<NGROUPS / 8, 256, 0, stream>>>(h, ln1_g + l * HID, ln1_b + l * HID, ybuf, NGROUPS);
    launch_gemm(ybuf, qkv_w + (size_t)l * 3 * HID * HID, qkv_b + (size_t)l * 3 * HID,
                nullptr, qkvb, NGROUPS, 3 * HID, HID, nullptr, nullptr, 0, stream);
    attn_kernel<<<dim3(NHEADS, NGROUPS / 16), 32, 0, stream>>>(qkvb, attno, NGROUPS);
    launch_gemm(attno, fc_w + (size_t)l * HID * HID, fc_b + (size_t)l * HID,
                h, h, NGROUPS, HID, HID, nullptr, nullptr, 0, stream);
    ln_kernel<<<NGROUPS / 8, 256, 0, stream>>>(h, ln2_g + l * HID, ln2_b + l * HID, ybuf, NGROUPS);
    launch_gemm(ybuf, mlp_w1 + (size_t)l * 4 * HID * HID, mlp_b1 + (size_t)l * 4 * HID,
                nullptr, gbuf, NGROUPS, 4 * HID, HID, nullptr, nullptr, 1, stream);
    launch_gemm(gbuf, mlp_w2 + (size_t)l * HID * 4 * HID, mlp_b2 + (size_t)l * HID,
                h, h, NGROUPS, HID, 4 * HID, nullptr, nullptr, 0, stream);
  }

  // ---- heads + scatter to points ----
  ln_kernel<<<NGROUPS / 8, 256, 0, stream>>>(h, he_ln_g, he_ln_b, ybuf, NGROUPS);
  launch_gemm(ybuf, he_w1, he_b1, nullptr, t1, NGROUPS, 256, 128, nullptr, nullptr, 1, stream);
  launch_gemm(t1, he_w2, he_b2, nullptr, ebuf, NGROUPS, 8, 256, nullptr, nullptr, 0, stream);
  ln_kernel<<<NGROUPS / 8, 256, 0, stream>>>(h, hp_ln_g, hp_ln_b, ybuf, NGROUPS);
  launch_gemm(ybuf, hp_w1, hp_b1, nullptr, t1, NGROUPS, 256, 128, nullptr, nullptr, 1, stream);
  launch_gemm(t1, hp_w2, hp_b2, nullptr, pbuf, NGROUPS, 8, 256, nullptr, nullptr, 0, stream);
  gather_out_kernel<<<(NPOINTS + 255) / 256, 256, 0, stream>>>(nearest, ebuf, pbuf,
                                                               (float*)d_out);
}
